// AllinonePrompt_76192719831252
// MI455X (gfx1250) — matrix operand order repaired and checked by reading, not compiled, not run on hardware
//
#include <hip/hip_runtime.h>
#include <hip/hip_bf16.h>

typedef __attribute__((ext_vector_type(16))) _Float16 v16h;
typedef __attribute__((ext_vector_type(8)))  _Float16 v8h;
typedef __attribute__((ext_vector_type(8)))  float    v8f;

#define DDIM 128
#define TTOK 64
#define THRESHOLD 0.2f
#define TOKH_STRIDE 136   // halves per row of tok_h  (128 + 8 pad)
#define TOKT_STRIDE 72    // halves per row of tokT_h (64 + 8 pad)
#define WH_STRIDE   72    // halves per row of per-wave weight tile
#define OUT_STRIDE  132   // floats per row of per-wave output stage (128 + 4 pad)
#define TPW 8             // tiles per wave

// Load 16 halves (two 16B-aligned LDS b128s) into a v16h.
__device__ __forceinline__ v16h load16h(const _Float16* p) {
    union { v16h v; v8h h[2]; } u;
    u.h[0] = *(const v8h*)(p);
    u.h[1] = *(const v8h*)(p + 16);
    return u.v;
}

__global__ __launch_bounds__(256)
void prompt_token_kernel(const float* __restrict__ x,
                         const float* __restrict__ tokens,
                         float* __restrict__ out,
                         int ntiles) {
    __shared__ _Float16 tok_h [TTOK * TOKH_STRIDE];   // tokens, row-major, f16
    __shared__ _Float16 tokT_h[DDIM * TOKT_STRIDE];   // tokens, transposed, f16
    __shared__ _Float16 wh    [8 * 16 * WH_STRIDE];   // per-wave masked weights, f16
    extern __shared__ float outs_all[];               // 8 * 16 * OUT_STRIDE floats (dynamic)

    const int tid = threadIdx.x;

    // ---- Stage tokens into LDS in f16, both orientations (once per block) ----
    {
        const int t = tid >> 2;
        const int q = tid & 3;
        const float* src = tokens + t * DDIM + q * 32;
        #pragma unroll
        for (int j = 0; j < 32; j += 4) {
            float4 f = *(const float4*)(src + j);
            const int c = q * 32 + j;
            _Float16 h0 = (_Float16)f.x, h1 = (_Float16)f.y;
            _Float16 h2 = (_Float16)f.z, h3 = (_Float16)f.w;
            tok_h[t * TOKH_STRIDE + c + 0] = h0;
            tok_h[t * TOKH_STRIDE + c + 1] = h1;
            tok_h[t * TOKH_STRIDE + c + 2] = h2;
            tok_h[t * TOKH_STRIDE + c + 3] = h3;
            tokT_h[(c + 0) * TOKT_STRIDE + t] = h0;
            tokT_h[(c + 1) * TOKT_STRIDE + t] = h1;
            tokT_h[(c + 2) * TOKT_STRIDE + t] = h2;
            tokT_h[(c + 3) * TOKT_STRIDE + t] = h3;
        }
    }
    __syncthreads();

    const int wave = tid >> 5;
    const int lane = tid & 31;
    const int m    = lane & 15;               // row (A) / column (B) index within tile
    const int hi8  = (lane >= 16) ? 8  : 0;   // A-fragment K base / C-row base, upper half-wave
    const int kofB = (lane >= 16) ? 16 : 0;   // unused; B fragments use hi8 as K base
    _Float16* mywh = wh + wave * 16 * WH_STRIDE;
    float*    outs = outs_all + wave * 16 * OUT_STRIDE;

    const int tile0 = blockIdx.x * (8 * TPW) + wave * TPW;  // 8 consecutive tiles per wave

    for (int it = 0; it < TPW; ++it) {
        const int tile = tile0 + it;
        if (tile >= ntiles) break;
        const int row_base = tile * 16;

        // ---- prefetch next tile's x region (contiguous 8KB = 128 lines, 4/lane) ----
        if (it + 1 < TPW && tile + 1 < ntiles) {
            const char* pn = (const char*)(x + (size_t)(tile + 1) * 16 * DDIM);
            #pragma unroll
            for (int j = 0; j < 4; ++j)
                __builtin_prefetch(pn + (size_t)(j * 32 + lane) * 64, 0, 3);
        }

        // ============== GEMM1: W[16x64] = X[16x128] * tokens^T ==============
        v8f Wacc[4] = {};
        const float* xrow = x + (size_t)(row_base + m) * DDIM;

        #pragma unroll
        for (int kc = 0; kc < 4; ++kc) {      // K = 128 in chunks of 32
            // A fragment (f32 -> f16): K(e) = hi8 + (e&7) + 16*(e>>3)
            v16h a;
            {
                const float* p = xrow + kc * 32 + hi8;
                float4 f0 = *(const float4*)(p);
                float4 f1 = *(const float4*)(p + 4);
                float4 f2 = *(const float4*)(p + 16);
                float4 f3 = *(const float4*)(p + 20);
                a[0]=(_Float16)f0.x;  a[1]=(_Float16)f0.y;  a[2]=(_Float16)f0.z;  a[3]=(_Float16)f0.w;
                a[4]=(_Float16)f1.x;  a[5]=(_Float16)f1.y;  a[6]=(_Float16)f1.z;  a[7]=(_Float16)f1.w;
                a[8]=(_Float16)f2.x;  a[9]=(_Float16)f2.y;  a[10]=(_Float16)f2.z; a[11]=(_Float16)f2.w;
                a[12]=(_Float16)f3.x; a[13]=(_Float16)f3.y; a[14]=(_Float16)f3.z; a[15]=(_Float16)f3.w;
            }
            // Batch all 4 B fragments first so LDS latency overlaps the WMMA group.
            v16h barr[4];
            #pragma unroll
            for (int tt = 0; tt < 4; ++tt)
                barr[tt] = load16h(tok_h + (tt * 16 + m) * TOKH_STRIDE + kc * 32 + hi8);
            #pragma unroll
            for (int tt = 0; tt < 4; ++tt)
                Wacc[tt] = __builtin_amdgcn_wmma_f32_16x16x32_f16(
                    false, a, false, barr[tt], (short)0, Wacc[tt], false, false);
        }

        // ---- sigmoid + threshold, write masked W to LDS in WMMA-A layout ----
        #pragma unroll
        for (int tt = 0; tt < 4; ++tt) {
            #pragma unroll
            for (int v = 0; v < 8; ++v) {
                float z = Wacc[tt][v];
                float w = 1.0f / (1.0f + __expf(-z));
                w = (w < THRESHOLD) ? 0.0f : w;
                mywh[(v + hi8) * WH_STRIDE + tt * 16 + m] = (_Float16)w;
            }
        }

        // ============== GEMM2: Y[16x128] = W[16x64] * tokens ==============
        v8f Yacc[8] = {};
        #pragma unroll
        for (int tc = 0; tc < 2; ++tc) {      // K = 64 tokens in chunks of 32
            v16h a;
            {
                const _Float16* ap = mywh + m * WH_STRIDE + tc * 32 + hi8;
                union { v16h v; v8h h[2]; } u;
                u.h[0] = *(const v8h*)(ap);
                u.h[1] = *(const v8h*)(ap + 16);
                a = u.v;
            }
            // Batch all 8 B fragments first (64 VGPRs), then the 8 WMMAs.
            v16h bb[8];
            #pragma unroll
            for (int ct = 0; ct < 8; ++ct)
                bb[ct] = load16h(tokT_h + (ct * 16 + m) * TOKT_STRIDE + tc * 32 + hi8);
            #pragma unroll
            for (int ct = 0; ct < 8; ++ct)
                Yacc[ct] = __builtin_amdgcn_wmma_f32_16x16x32_f16(
                    false, a, false, bb[ct], (short)0, Yacc[ct], false, false);
        }

        // ---- stage Y to LDS (conflict-free), then coalesced out = x + Y ----
        #pragma unroll
        for (int ct = 0; ct < 8; ++ct) {
            #pragma unroll
            for (int v = 0; v < 8; ++v)
                outs[(v + hi8) * OUT_STRIDE + ct * 16 + m] = Yacc[ct][v];
        }
        const float* xt = x   + (size_t)row_base * DDIM;
        float*       ot = out + (size_t)row_base * DDIM;
        #pragma unroll
        for (int r = 0; r < 16; ++r) {
            float4 xf = *(const float4*)(xt   + r * DDIM + lane * 4);
            float4 yf = *(const float4*)(outs + r * OUT_STRIDE + lane * 4);
            float4 o;
            o.x = xf.x + yf.x;  o.y = xf.y + yf.y;
            o.z = xf.z + yf.z;  o.w = xf.w + yf.w;
            *(float4*)(ot + r * DDIM + lane * 4) = o;
        }
    }
}

extern "C" void kernel_launch(void* const* d_in, const int* in_sizes, int n_in,
                              void* d_out, int out_size, void* d_ws, size_t ws_size,
                              hipStream_t stream) {
    const float* x      = (const float*)d_in[0];
    const float* tokens = (const float*)d_in[1];
    float* out          = (float*)d_out;
    const int N      = in_sizes[0] / DDIM;        // 1,000,000 (multiple of 16)
    const int ntiles = N / 16;                    // 62,500 row tiles
    const int tiles_per_block = 8 * TPW;          // 64
    const int blocks = (ntiles + tiles_per_block - 1) / tiles_per_block;
    const size_t dyn_lds = (size_t)8 * 16 * OUT_STRIDE * sizeof(float);  // 67,584 B
    hipLaunchKernelGGL(prompt_token_kernel, dim3(blocks), dim3(256), dyn_lds, stream,
                       x, tokens, out, ntiles);
}
